// LoRAExpertMLP_4131758539046
// MI455X (gfx1250) — matrix operand order repaired and checked by reading, not compiled, not run on hardware
//
#include <hip/hip_runtime.h>
#include <hip/hip_bf16.h>

typedef __bf16 bf16;
typedef __attribute__((ext_vector_type(8)))  bf16  bf16x8;
typedef __attribute__((ext_vector_type(16))) bf16  v16bf;
typedef __attribute__((ext_vector_type(8)))  float v8f;

namespace {
constexpr int kB = 4, kS = 2048, kD = 2048, kDFF = 5632, kE = 8, kR = 16;
constexpr int kN   = kB * kS;    // 8192 tokens
constexpr int kGUW = 2 * kDFF;   // 11264 (gate cols then up cols)
constexpr float kScale = 2.0f;   // alpha / rank = 32/16
}

__device__ __forceinline__ float silu_f(float x) {
  return x / (1.0f + __expf(-x));
}

// ---------------------------------------------------------------------------
// CDNA5 async global->LDS copy (ASYNCcnt-tracked, bypasses VGPRs).
// LDS destination is the low 32 bits of the generic shared-memory pointer
// (ISA: flat LDS aperture, LDS offset is the low 32 address bits).
// ---------------------------------------------------------------------------
__device__ __forceinline__ void async_load_b128(void* lds_dst, const void* gsrc) {
  unsigned lds_off = (unsigned)(unsigned long long)(size_t)lds_dst;
  unsigned long long ga = (unsigned long long)(size_t)gsrc;
  asm volatile("global_load_async_to_lds_b128 %0, %1, off"
               :
               : "v"(lds_off), "v"(ga)
               : "memory");
}

__device__ __forceinline__ void wait_async0() {
#if __has_builtin(__builtin_amdgcn_s_wait_asynccnt)
  __builtin_amdgcn_s_wait_asynccnt(0);
#else
  asm volatile("s_wait_asynccnt 0x0" ::: "memory");
#endif
}

// ---------------------------------------------------------------------------
// fp32 -> bf16 conversion (grid-stride)
// ---------------------------------------------------------------------------
__global__ void cvt_kernel(const float* __restrict__ src, bf16* __restrict__ dst,
                           size_t n) {
  size_t i = (size_t)blockIdx.x * blockDim.x + threadIdx.x;
  const size_t stride = (size_t)gridDim.x * blockDim.x;
  for (; i < n; i += stride) dst[i] = (bf16)src[i];
}

// ---------------------------------------------------------------------------
// Tiled WMMA GEMM: C[M,Ncols] = A[M,K] * Bw[Ncols,K]^T   (bf16 in, fp32 acc)
// Block: 256 threads (8 waves, 2x4), tile 128x128, K-step 32.
// Double-buffered LDS; B (and A when not fused) staged with async-to-LDS,
// one barrier per K-tile:
//   wait_asynccnt(0); barrier; issue tile k+1 -> buf[1-c]; wmma from buf[c]
// FUSE_SILU: A points at the [N,2*DFF] gate/up buffer; the A-stager computes
//            silu(gate)*up on the fly (lda = 2*DFF, up offset = lda/2).
// STORE_BF16: write C as bf16 (gate/up GEMM) else fp32 (down GEMM).
// ---------------------------------------------------------------------------
template <int STORE_BF16, int FUSE_SILU>
__global__ __launch_bounds__(256) void wmma_gemm_kernel(
    const bf16* __restrict__ A, const bf16* __restrict__ Bw,
    void* __restrict__ Cout, int M, int Ncols, int K, int lda, int ldc) {
  (void)M; (void)Ncols;
  constexpr int LSTR = 40;        // bf16 per LDS row (80B step, conflict-free)
  constexpr int BUFE = 128 * LSTR;
  __shared__ bf16 As[2 * BUFE];   // ping-pong
  __shared__ bf16 Bs[2 * BUFE];

  const int tid  = threadIdx.x;
  const int lane = tid & 31;
  const int wid  = tid >> 5;
  const int half = lane >> 4;
  const int l16  = lane & 15;
  const int wm   = wid >> 2;  // 0..1 : 64 rows each
  const int wn   = wid & 3;   // 0..3 : 32 cols each
  const int m0   = blockIdx.y * 128;
  const int n0   = blockIdx.x * 128;

  v8f acc[4][2];
#pragma unroll
  for (int mi = 0; mi < 4; ++mi)
#pragma unroll
    for (int ni = 0; ni < 2; ++ni)
#pragma unroll
      for (int j = 0; j < 8; ++j) acc[mi][ni][j] = 0.0f;

  const int sr  = tid >> 1;        // staging row/col 0..127
  const int skp = (tid & 1) * 16;  // k sub-chunk 0 or 16

  auto issue_tile = [&](int kt, int buf) {
    const int k0 = kt * 32;
    // ---- B tile (always async) ----
    {
      const bf16* src = Bw + (size_t)(n0 + sr) * K + (k0 + skp);
      bf16* dst = &Bs[buf * BUFE + sr * LSTR + skp];
      async_load_b128(dst, src);
      async_load_b128(dst + 8, src + 8);
    }
    // ---- A tile ----
    if (FUSE_SILU) {
      const bf16* gp = A + (size_t)(m0 + sr) * lda + (k0 + skp);
      const bf16x8* g8 = (const bf16x8*)gp;
      const bf16x8* u8 = (const bf16x8*)(gp + (lda >> 1));
      bf16* dst = &As[buf * BUFE + sr * LSTR + skp];
#pragma unroll
      for (int v = 0; v < 2; ++v) {
        bf16x8 g = g8[v], u = u8[v];
#pragma unroll
        for (int j = 0; j < 8; ++j) {
          float gf = (float)g[j];
          float uf = (float)u[j];
          dst[v * 8 + j] = (bf16)(silu_f(gf) * uf);
        }
      }
    } else {
      const bf16* src = A + (size_t)(m0 + sr) * lda + (k0 + skp);
      bf16* dst = &As[buf * BUFE + sr * LSTR + skp];
      async_load_b128(dst, src);
      async_load_b128(dst + 8, src + 8);
    }
  };

  const int ktiles = K / 32;
  issue_tile(0, 0);  // prologue

  for (int kt = 0; kt < ktiles; ++kt) {
    const int c = kt & 1;
    wait_async0();     // my slice of tile kt has landed in LDS
    __syncthreads();   // everyone's slice landed; buf[1-c] free to overwrite
    if (kt + 1 < ktiles) issue_tile(kt + 1, 1 - c);

    // ---- fragments (ISA §7.12.2 layouts) ----
    v16bf afr[4], bfr[2];
#pragma unroll
    for (int mi = 0; mi < 4; ++mi) {
      const bf16* ap = &As[c * BUFE + (wm * 64 + mi * 16 + l16) * LSTR + half * 8];
#pragma unroll
      for (int j = 0; j < 8; ++j) {
        afr[mi][j]     = ap[j];       // K = half*8 + j
        afr[mi][8 + j] = ap[16 + j];  // K = 16 + half*8 + j
      }
    }
#pragma unroll
    for (int ni = 0; ni < 2; ++ni) {
      const bf16* bp = &Bs[c * BUFE + (wn * 32 + ni * 16 + l16) * LSTR + half * 8];
#pragma unroll
      for (int j = 0; j < 8; ++j) {
        bfr[ni][j]     = bp[j];       // K = half*8 + j
        bfr[ni][8 + j] = bp[16 + j];  // K = 16 + half*8 + j
      }
    }
#pragma unroll
    for (int mi = 0; mi < 4; ++mi)
#pragma unroll
      for (int ni = 0; ni < 2; ++ni)
        acc[mi][ni] = __builtin_amdgcn_wmma_f32_16x16x32_bf16(
            false, afr[mi], false, bfr[ni], (short)0, acc[mi][ni], false, false);
  }

  // ---- epilogue: C layout VGPR i -> M = i + 8*half, N = lane%16 ----
#pragma unroll
  for (int mi = 0; mi < 4; ++mi)
#pragma unroll
    for (int ni = 0; ni < 2; ++ni)
#pragma unroll
      for (int i = 0; i < 8; ++i) {
        int row = m0 + wm * 64 + mi * 16 + i + 8 * half;
        int col = n0 + wn * 32 + ni * 16 + l16;
        if (STORE_BF16)
          ((bf16*)Cout)[(size_t)row * ldc + col] = (bf16)acc[mi][ni][i];
        else
          ((float*)Cout)[(size_t)row * ldc + col] = acc[mi][ni][i];
      }
}

// ---------------------------------------------------------------------------
// Router: one wave per token; fp32 logits, softmax, top-2 -> expert_w [N,E]
// ---------------------------------------------------------------------------
__global__ __launch_bounds__(256) void router_kernel(
    const float* __restrict__ x, const float* __restrict__ Wr,
    float* __restrict__ ew) {
  const int lane = threadIdx.x & 31;
  const int wid  = threadIdx.x >> 5;
  const int n = blockIdx.x * 8 + wid;
  const float* xr = x + (size_t)n * kD;
  float logit[kE];
#pragma unroll 1
  for (int e = 0; e < kE; ++e) {
    const float* wr = Wr + (size_t)e * kD;
    float p = 0.0f;
    for (int d = lane; d < kD; d += 32) p += xr[d] * wr[d];
#pragma unroll
    for (int off = 16; off > 0; off >>= 1) p += __shfl_xor(p, off, 32);
    logit[e] = p;
  }
  float mx = logit[0];
#pragma unroll
  for (int e = 1; e < kE; ++e) mx = fmaxf(mx, logit[e]);
  float pr[kE];
  float den = 0.0f;
#pragma unroll
  for (int e = 0; e < kE; ++e) { pr[e] = __expf(logit[e] - mx); den += pr[e]; }
  const float inv = 1.0f / den;
#pragma unroll
  for (int e = 0; e < kE; ++e) pr[e] *= inv;
  int i1 = 0;
#pragma unroll
  for (int e = 1; e < kE; ++e) if (pr[e] > pr[i1]) i1 = e;
  int i2 = (i1 == 0) ? 1 : 0;
#pragma unroll
  for (int e = 0; e < kE; ++e) if (e != i1 && pr[e] > pr[i2]) i2 = e;
  if (lane < kE)
    ew[(size_t)n * kE + lane] = (lane == i1 || lane == i2) ? pr[lane] : 0.0f;
}

// ---------------------------------------------------------------------------
// LoRA A projections via WMMA: t[gz][e][n][r] = SCALE * sum_k x[n,k]*A[e][r,k]
// One wave per (16-token tile, expert, gate/up); K loop over D.
// ---------------------------------------------------------------------------
__global__ __launch_bounds__(32) void lora_t_kernel(
    const bf16* __restrict__ xbf, const bf16* __restrict__ Abf,
    float* __restrict__ t_buf) {
  const int lane = threadIdx.x & 31;
  const int half = lane >> 4;
  const int l16  = lane & 15;
  const int e  = blockIdx.y;
  const int gz = blockIdx.z;
  const int t0 = blockIdx.x * 16;

  const bf16* xrow = xbf + (size_t)(t0 + l16) * kD;
  const bf16* arow =
      Abf + ((size_t)gz * kE + e) * (size_t)kR * kD + (size_t)l16 * kD;

  v8f acc = {0.f, 0.f, 0.f, 0.f, 0.f, 0.f, 0.f, 0.f};
  for (int k0 = 0; k0 < kD; k0 += 32) {
    v16bf a, b;
    const bf16* ap = xrow + k0 + half * 8;
#pragma unroll
    for (int j = 0; j < 8; ++j) { a[j] = ap[j]; a[8 + j] = ap[16 + j]; }
    const bf16* bp = arow + k0 + half * 8;
#pragma unroll
    for (int j = 0; j < 8; ++j) { b[j] = bp[j]; b[8 + j] = bp[16 + j]; }
    acc = __builtin_amdgcn_wmma_f32_16x16x32_bf16(false, a, false, b, (short)0,
                                                  acc, false, false);
  }
  float* to = t_buf + (((size_t)gz * kE + e) * kN + t0) * kR;
#pragma unroll
  for (int i = 0; i < 8; ++i)
    to[(size_t)(i + 8 * half) * kR + l16] = acc[i] * kScale;
}

// ---------------------------------------------------------------------------
// Fused expert pass: per 16-token tile, per 64-wide DFF chunk: compute ALL 8
// experts' h_e = silu(g + t_g@Bg^T) * (u + t_u@Bu^T) into LDS (32 KB), one
// barrier, then reduce s_e[n][r] += sum_f h_e[n][f] * Ad[e][r][f] for all
// experts, one barrier.  h_e is never materialized in global memory.
// ---------------------------------------------------------------------------
__global__ __launch_bounds__(256) void expert_kernel(
    const bf16* __restrict__ gu, const float* __restrict__ t_buf,
    const float* __restrict__ Bg, const float* __restrict__ Bu,
    const float* __restrict__ Ad, float* __restrict__ s_buf) {
  __shared__ float tg[kE][16][kR];    // 8 KB (pre-scaled by kScale)
  __shared__ float tu[kE][16][kR];    // 8 KB
  __shared__ float hsh[kE][16][64];   // 32 KB
  const int tid = threadIdx.x;
  const int t0  = blockIdx.x * 16;

  for (int i = tid; i < kE * 16 * kR; i += 256) {
    int e = i >> 8, n = (i >> 4) & 15, r = i & 15;
    tg[e][n][r] = t_buf[((size_t)e * kN + t0 + n) * kR + r];
    tu[e][n][r] = t_buf[(((size_t)kE + e) * kN + t0 + n) * kR + r];
  }
  __syncthreads();

  float sacc[kE];
#pragma unroll
  for (int e = 0; e < kE; ++e) sacc[e] = 0.0f;

  const int n4  = tid >> 6;  // stage1: 4 tokens per thread
  const int fl  = tid & 63;  // stage1: f within chunk
  const int sn  = tid & 15;  // stage2: token
  const int srr = tid >> 4;  // stage2: rank

  for (int f0 = 0; f0 < kDFF; f0 += 64) {
    const int f = f0 + fl;
    float gv[4], uv[4];
#pragma unroll
    for (int i = 0; i < 4; ++i) {
      const size_t rowb = (size_t)(t0 + n4 * 4 + i) * kGUW;
      gv[i] = (float)gu[rowb + f];
      uv[i] = (float)gu[rowb + kDFF + f];
    }
    // stage 1: all experts' h into LDS
#pragma unroll 1
    for (int e = 0; e < kE; ++e) {
      const float* bg = Bg + ((size_t)e * kDFF + f) * kR;
      const float* bu = Bu + ((size_t)e * kDFF + f) * kR;
#pragma unroll
      for (int i = 0; i < 4; ++i) {
        const int n = n4 * 4 + i;
        float dg = 0.0f, du = 0.0f;
#pragma unroll
        for (int r = 0; r < kR; ++r) {
          dg += tg[e][n][r] * bg[r];
          du += tu[e][n][r] * bu[r];
        }
        hsh[e][n][fl] = silu_f(gv[i] + dg) * (uv[i] + du);
      }
    }
    __syncthreads();
    // stage 2: rank-16 reduction for all experts
#pragma unroll 1
    for (int e = 0; e < kE; ++e) {
      const float* ad = Ad + ((size_t)e * kR + srr) * kDFF + f0;
      float a = 0.0f;
#pragma unroll
      for (int j = 0; j < 64; ++j) a += hsh[e][sn][j] * ad[j];
      sacc[e] += a;
    }
    __syncthreads();
  }
#pragma unroll
  for (int e = 0; e < kE; ++e)
    s_buf[((size_t)e * kN + t0 + sn) * kR + srr] = sacc[e];
}

// ---------------------------------------------------------------------------
// Combine: out[n][d] += sum_e kScale * w[n][e] * sum_r s[e][n][r]*Bd[e][d][r]
// ---------------------------------------------------------------------------
__global__ __launch_bounds__(256) void combine_kernel(
    const float* __restrict__ s_buf, const float* __restrict__ Bd,
    const float* __restrict__ ew, float* __restrict__ out) {
  __shared__ float sw[kE][16][kR];
  const int tid = threadIdx.x;
  const int t0  = blockIdx.x * 16;
  for (int i = tid; i < kE * 16 * kR; i += 256) {
    int e = i >> 8, n = (i >> 4) & 15, r = i & 15;
    sw[e][n][r] = s_buf[((size_t)e * kN + t0 + n) * kR + r] * kScale *
                  ew[(size_t)(t0 + n) * kE + e];
  }
  __syncthreads();
  const int n  = tid & 15;
  const int dl = tid >> 4;
  for (int d0 = 0; d0 < kD; d0 += 16) {
    const int d = d0 + dl;
    float acc = out[(size_t)(t0 + n) * kD + d];
#pragma unroll 1
    for (int e = 0; e < kE; ++e) {
      const float* bd = Bd + ((size_t)e * kD + d) * kR;
#pragma unroll
      for (int r = 0; r < kR; ++r) acc += sw[e][n][r] * bd[r];
    }
    out[(size_t)(t0 + n) * kD + d] = acc;
  }
}

// ---------------------------------------------------------------------------
extern "C" void kernel_launch(void* const* d_in, const int* in_sizes, int n_in,
                              void* d_out, int out_size, void* d_ws,
                              size_t ws_size, hipStream_t stream) {
  (void)in_sizes; (void)n_in; (void)out_size; (void)ws_size;
  const float* x  = (const float*)d_in[0];
  const float* Wg = (const float*)d_in[1];
  const float* Wu = (const float*)d_in[2];
  const float* Wd = (const float*)d_in[3];
  const float* Wr = (const float*)d_in[4];
  const float* Ag = (const float*)d_in[5];
  const float* Bg = (const float*)d_in[6];
  const float* Au = (const float*)d_in[7];
  const float* Bu = (const float*)d_in[8];
  const float* Ad = (const float*)d_in[9];
  const float* Bd = (const float*)d_in[10];
  float* out = (float*)d_out;

  char* ws = (char*)d_ws;
  size_t off = 0;
  auto take = [&](size_t bytes) -> char* {
    char* p = ws + off;
    off += (bytes + 255) & ~(size_t)255;
    return p;
  };
  bf16*  x_bf   = (bf16*)take((size_t)kN * kD * 2);
  bf16*  Wgu_bf = (bf16*)take((size_t)kGUW * kD * 2);
  bf16*  Wd_bf  = (bf16*)take((size_t)kD * kDFF * 2);
  bf16*  A_bf   = (bf16*)take((size_t)2 * kE * kR * kD * 2);
  bf16*  gubuf  = (bf16*)take((size_t)kN * kGUW * 2);
  float* t_buf  = (float*)take((size_t)2 * kE * kN * kR * 4);
  float* s_buf  = (float*)take((size_t)kE * kN * kR * 4);
  float* ewbuf  = (float*)take((size_t)kN * kE * 4);

  // 1) fp32 -> bf16 conversions
  cvt_kernel<<<2048, 256, 0, stream>>>(x, x_bf, (size_t)kN * kD);
  cvt_kernel<<<2048, 256, 0, stream>>>(Wg, Wgu_bf, (size_t)kDFF * kD);
  cvt_kernel<<<2048, 256, 0, stream>>>(Wu, Wgu_bf + (size_t)kDFF * kD,
                                       (size_t)kDFF * kD);
  cvt_kernel<<<2048, 256, 0, stream>>>(Wd, Wd_bf, (size_t)kD * kDFF);
  cvt_kernel<<<256, 256, 0, stream>>>(Ag, A_bf, (size_t)kE * kR * kD);
  cvt_kernel<<<256, 256, 0, stream>>>(Au, A_bf + (size_t)kE * kR * kD,
                                      (size_t)kE * kR * kD);

  // 2) gate/up GEMM: [N,D] x [D,2*DFF] -> bf16 gu buffer
  wmma_gemm_kernel<1, 0><<<dim3(kGUW / 128, kN / 128), 256, 0, stream>>>(
      x_bf, Wgu_bf, gubuf, kN, kGUW, kD, kD, kGUW);

  // 3) router (fp32 exact)
  router_kernel<<<kN / 8, 256, 0, stream>>>(x, Wr, ewbuf);

  // 4) LoRA-A projections t = x @ A^T (WMMA, pre-scaled)
  lora_t_kernel<<<dim3(kN / 16, kE, 2), 32, 0, stream>>>(x_bf, A_bf, t_buf);

  // 5) fused expert pass -> s_e[n][r]
  expert_kernel<<<kN / 16, 256, 0, stream>>>(gubuf, t_buf, Bg, Bu, Ad, s_buf);

  // 6) down GEMM with fused silu(g)*u A-stager: -> fp32 d_out
  wmma_gemm_kernel<0, 1><<<dim3(kD / 128, kN / 128), 256, 0, stream>>>(
      gubuf, Wd_bf, out, kN, kD, kDFF, kGUW, kD);

  // 7) add expert deltas into d_out
  combine_kernel<<<kN / 16, 256, 0, stream>>>(s_buf, Bd, ewbuf, out);
}
